// MultiHeadAttention_6511170421555
// MI455X (gfx1250) — compile-verified, matrix operand order checked by reading, not run on hardware
//
#include <hip/hip_runtime.h>
#include <cstddef>
#include <cstdint>

// ---------------------------------------------------------------------------
// CDNA5 (gfx1250) attention block: QKV GEMM -> RoPE -> flash attention -> O GEMM
// Matmuls: v_wmma_f32_16x16x32_bf16. Flash-attn K tiles staged via the Tensor
// Data Mover (tensor_load_to_lds + s_wait_tensorcnt) when available.
// ---------------------------------------------------------------------------

typedef __attribute__((ext_vector_type(8)))  float        v8f;
typedef __attribute__((ext_vector_type(4)))  __bf16       v4bf;
typedef __attribute__((ext_vector_type(8)))  __bf16       v8bf;
typedef __attribute__((ext_vector_type(16))) __bf16       v16bf;
typedef __attribute__((ext_vector_type(4)))  unsigned int v4u;
typedef __attribute__((ext_vector_type(4)))  int          v4i;
typedef __attribute__((ext_vector_type(8)))  int          v8i;

#define CAT16(lo, hi) __builtin_shufflevector((lo), (hi), 0,1,2,3,4,5,6,7,8,9,10,11,12,13,14,15)

#if defined(__HIP_DEVICE_COMPILE__) && __has_builtin(__builtin_amdgcn_tensor_load_to_lds)
#define USE_TDM 1
#else
#define USE_TDM 0
#endif

constexpr int kB   = 2;
constexpr int kS   = 2048;
constexpr int kHid = 2048;
constexpr int kNH  = 16;
constexpr int kNKV = 4;
constexpr int kHD  = 128;
constexpr int kQDim  = kNH * kHD;            // 2048
constexpr int kKVDim = kNKV * kHD;           // 512
constexpr int kQKVD  = kQDim + 2 * kKVDim;   // 3072
constexpr int kMRows = kB * kS;              // 4096
constexpr float kRopeBase = 10000.0f;

// ---------------------------------------------------------------------------
// GEMM: C[M,N] = A[M,K] * Bmat[N,K]^T   (Bmat row-major N x K, as w_qkv / w_o)
// Block tile 128x128, K-step 32, 256 threads = 8 waves, each wave 64x32.
// ---------------------------------------------------------------------------
template<bool A_BF16>
__global__ __launch_bounds__(256)
void gemm_bf16_wmma(const void* __restrict__ Aptr,
                    const float* __restrict__ Bmat,
                    float* __restrict__ C,
                    int M, int N, int K)
{
    constexpr int LS = 40;                    // LDS row stride (bf16): 80B, 16B aligned
    __shared__ __bf16 As[128 * LS];
    __shared__ __bf16 Bs[128 * LS];

    const int tid  = threadIdx.x;
    const int lane = tid & 31;
    const int wave = tid >> 5;                // 0..7
    const int lr   = lane & 15;
    const int kg   = lane >> 4;               // 0/1
    const int waveM = (wave >> 2) * 64;       // 0 or 64
    const int waveN = (wave & 3) * 32;        // 0..96
    const int blockM = blockIdx.y * 128;
    const int blockN = blockIdx.x * 128;

    v8f acc[4][2];
#pragma unroll
    for (int i = 0; i < 4; ++i)
#pragma unroll
        for (int j = 0; j < 2; ++j)
            acc[i][j] = v8f{};

    for (int k0 = 0; k0 < K; k0 += 32) {
        // L2 prefetch of the next B tile (lowered to global_prefetch_b8).
        if (k0 + 32 < K)
            __builtin_prefetch(Bmat + (size_t)(blockN + (tid >> 1)) * K +
                               (k0 + 32) + (tid & 1) * 16, 0, 1);
        __syncthreads();
        // ---- stage A tile (128 x 32) into LDS as bf16 ----
        if constexpr (!A_BF16) {
            const float* A = (const float*)Aptr;
#pragma unroll
            for (int i = 0; i < 4; ++i) {
                int idx = tid + i * 256;          // float4 index (1024 total)
                int row = idx >> 3;
                int c4  = (idx & 7) << 2;
                float4 f = *(const float4*)(A + (size_t)(blockM + row) * K + k0 + c4);
                v4bf pk = { (__bf16)f.x, (__bf16)f.y, (__bf16)f.z, (__bf16)f.w };
                *(v4bf*)&As[row * LS + c4] = pk;  // one ds_store_b64
            }
        } else {
            const __bf16* A = (const __bf16*)Aptr;
#pragma unroll
            for (int i = 0; i < 2; ++i) {
                int idx = tid + i * 256;          // 8-elem chunk index (512 total)
                int row = idx >> 2;
                int c8  = (idx & 3) << 3;
                v8bf v = *(const v8bf*)(A + (size_t)(blockM + row) * K + k0 + c8);
                *(v8bf*)&As[row * LS + c8] = v;
            }
        }
        // ---- stage B tile (128 rows of Bmat x 32 cols) into LDS as bf16 ----
#pragma unroll
        for (int i = 0; i < 4; ++i) {
            int idx = tid + i * 256;
            int row = idx >> 3;
            int c4  = (idx & 7) << 2;
            float4 f = *(const float4*)(Bmat + (size_t)(blockN + row) * K + k0 + c4);
            v4bf pk = { (__bf16)f.x, (__bf16)f.y, (__bf16)f.z, (__bf16)f.w };
            *(v4bf*)&Bs[row * LS + c4] = pk;
        }
        __syncthreads();

        // ---- B fragments (2 N-subtiles of 16) ----
        v16bf bfrag[2];
#pragma unroll
        for (int nj = 0; nj < 2; ++nj) {
            const __bf16* p = &Bs[(waveN + nj * 16 + lr) * LS + kg * 8];
            v8bf lo = *(const v8bf*)p;
            v8bf hi = *(const v8bf*)(p + 16);
            bfrag[nj] = CAT16(lo, hi);
        }
        // ---- A fragments + WMMA ----
#pragma unroll
        for (int mi = 0; mi < 4; ++mi) {
            const __bf16* p = &As[(waveM + mi * 16 + lr) * LS + kg * 8];
            v8bf lo = *(const v8bf*)p;
            v8bf hi = *(const v8bf*)(p + 16);
            v16bf afrag = CAT16(lo, hi);
#pragma unroll
            for (int nj = 0; nj < 2; ++nj) {
                acc[mi][nj] = __builtin_amdgcn_wmma_f32_16x16x32_bf16(
                    false, afrag, false, bfrag[nj], (short)0, acc[mi][nj], false, false);
            }
        }
    }

    // ---- epilogue: C layout: row = base + j + 8*kg, col = base + lr ----
#pragma unroll
    for (int mi = 0; mi < 4; ++mi)
#pragma unroll
        for (int nj = 0; nj < 2; ++nj) {
            int col = blockN + waveN + nj * 16 + lr;
            int rowb = blockM + waveM + mi * 16 + kg * 8;
#pragma unroll
            for (int j = 0; j < 8; ++j)
                C[(size_t)(rowb + j) * N + col] = acc[mi][nj][j];
        }
}

// ---------------------------------------------------------------------------
// RoPE + split qkv(f32, [B*S, 3072]) into head-major bf16 q/k/v.
//   q: (B, NH, S, HD)  k,v: (B, NKV, S, HD)
// ---------------------------------------------------------------------------
__global__ __launch_bounds__(256)
void rope_split_kernel(const float* __restrict__ qkv,
                       __bf16* __restrict__ q,
                       __bf16* __restrict__ k,
                       __bf16* __restrict__ v)
{
    const int bs = blockIdx.x;               // 0 .. B*S-1
    const int b  = bs / kS;
    const int s  = bs % kS;
    const float* row = qkv + (size_t)bs * kQKVD;
    const int t = threadIdx.x;
    const float fs = (float)s;

    // q heads
    for (int p = t; p < kNH * 64; p += 256) {
        int h = p >> 6, d = p & 63;
        float inv = __powf(kRopeBase, -(float)d / 64.0f);
        float fr  = fs * inv;
        float c = __cosf(fr), sn = __sinf(fr);
        float x1 = row[h * kHD + d];
        float x2 = row[h * kHD + d + 64];
        size_t o = ((size_t)(b * kNH + h) * kS + s) * kHD;
        q[o + d]      = (__bf16)(x1 * c - x2 * sn);
        q[o + d + 64] = (__bf16)(x2 * c + x1 * sn);
    }
    // k heads
    for (int p = t; p < kNKV * 64; p += 256) {
        int h = p >> 6, d = p & 63;
        float inv = __powf(kRopeBase, -(float)d / 64.0f);
        float fr  = fs * inv;
        float c = __cosf(fr), sn = __sinf(fr);
        float x1 = row[kQDim + h * kHD + d];
        float x2 = row[kQDim + h * kHD + d + 64];
        size_t o = ((size_t)(b * kNKV + h) * kS + s) * kHD;
        k[o + d]      = (__bf16)(x1 * c - x2 * sn);
        k[o + d + 64] = (__bf16)(x2 * c + x1 * sn);
    }
    // v copy
    for (int e = t; e < kKVDim; e += 256) {
        int h = e >> 7, d = e & 127;
        v[((size_t)(b * kNKV + h) * kS + s) * kHD + d] =
            (__bf16)row[kQDim + kKVDim + e];
    }
}

// ---------------------------------------------------------------------------
// Flash attention (causal, GQA 16q/4kv). One workgroup = (b, h, 128 q rows);
// 8 waves, each owns 16 q rows. Key blocks of 32, online softmax.
// K tile staged by the Tensor Data Mover (D# with LDS padding to stride 136),
// V tile staged transposed through VGPRs.
// ---------------------------------------------------------------------------
__global__ __launch_bounds__(256)
void flash_attn_kernel(const __bf16* __restrict__ Q,
                       const __bf16* __restrict__ Kg,
                       const __bf16* __restrict__ Vg,
                       __bf16* __restrict__ O)
{
    constexpr int KST = 136;                 // Ks row stride (bf16) = 128 + 8 pad
    constexpr int VST = 40;                  // Vt row stride
    constexpr int PST = 40;                  // Ps row stride
    __shared__ __bf16 Ks[32 * KST];          // [key][d]
    __shared__ __bf16 Vt[128 * VST];         // [d][key]   (transposed V)
    __shared__ __bf16 Ps[8][16 * PST];       // per-wave P tile [row][key]

    const int qb  = blockIdx.x;              // S/128
    const int h   = blockIdx.y;
    const int b   = blockIdx.z;
    const int kvh = h >> 2;                  // NH/NKV = 4

    const int tid  = threadIdx.x;
    const int lane = tid & 31;
    const int wave = tid >> 5;
    const int lr   = lane & 15;
    const int kg   = lane >> 4;

    const int qrow0 = qb * 128 + wave * 16;  // wave's first q row
    const __bf16* qbase = Q + ((size_t)(b * kNH + h) * kS + qrow0) * kHD;
    const __bf16* kbase = Kg + (size_t)(b * kNKV + kvh) * kS * kHD;
    const __bf16* vbase = Vg + (size_t)(b * kNKV + kvh) * kS * kHD;

    // Preload this wave's Q fragments (4 K-chunks of 32 over HD=128).
    v16bf qf[4];
#pragma unroll
    for (int c = 0; c < 4; ++c) {
        const __bf16* p = qbase + (size_t)lr * kHD + c * 32 + kg * 8;
        v8bf lo = *(const v8bf*)p;
        v8bf hi = *(const v8bf*)(p + 16);
        qf[c] = CAT16(lo, hi);
    }

    float m[8], l[8];
    v8f o[8];
#pragma unroll
    for (int j = 0; j < 8; ++j) { m[j] = -1e30f; l[j] = 0.0f; }
#pragma unroll
    for (int t2 = 0; t2 < 8; ++t2) o[t2] = v8f{};

    const float scale = 0.08838834764831845f;   // 1/sqrt(128)
    const int nkb = (qb + 1) * 4;               // causal key-block count

    for (int kb = 0; kb < nkb; ++kb) {
        const int key0 = kb * 32;
        __syncthreads();

        // ---- stage K block [32 keys][128 d] ----
#if USE_TDM
        if (wave == 0) {
            // Tensor DMA descriptor: 2D tile 128(d) x 32(keys), 2B elems,
            // LDS padding: +4 DWORDs every 64 DWORDs -> row stride 136 bf16.
            const unsigned ldsKs = (unsigned)(uintptr_t)&Ks[0];
            const uint64_t ga = (uint64_t)(uintptr_t)(kbase + (size_t)key0 * kHD);
            v4u g0 = { 1u,                                     // count=1, user D#
                       ldsKs,                                  // lds_addr
                       (unsigned)ga,                           // global_addr[31:0]
                       (unsigned)((ga >> 32) & 0x1FFFFFFull)   // global_addr[56:32]
                           | 0x80000000u };                    // type=2 ("image")
            v8i g1 = { (int)0x07510000,     // data_size=2B, pad_en, ivl=64dw, amt=4dw
                       (int)(128u << 16),   // tensor_dim0 = HD
                       (int)(2048u << 16),  // tensor_dim1 = S
                       (int)(128u << 16),   // tile_dim0 = 128
                       32,                  // tile_dim1 = 32 keys
                       128,                 // tensor_dim0_stride = HD
                       0, 0 };
            v4i z4 = { 0, 0, 0, 0 };
#if __clang_major__ >= 23
            v8i z8 = { 0, 0, 0, 0, 0, 0, 0, 0 };
            __builtin_amdgcn_tensor_load_to_lds(g0, g1, z4, z4, z8, 0);
#else
            __builtin_amdgcn_tensor_load_to_lds(g0, g1, z4, z4, 0);
#endif
#if __has_builtin(__builtin_amdgcn_s_wait_tensorcnt)
            __builtin_amdgcn_s_wait_tensorcnt(0);
#else
            asm volatile("s_wait_tensorcnt 0x0" ::: "memory");
#endif
        }
#else
#pragma unroll
        for (int i = 0; i < 2; ++i) {
            int idx = tid + i * 256;            // 512 chunks of 8 bf16
            int key = idx >> 4;
            int d8  = (idx & 15) << 3;
            v8bf val = *(const v8bf*)(kbase + (size_t)(key0 + key) * kHD + d8);
            *(v8bf*)&Ks[key * KST + d8] = val;
        }
#endif
        // ---- stage V block transposed: Vt[d][key] ----
#pragma unroll
        for (int i = 0; i < 2; ++i) {
            int idx = tid + i * 256;
            int key = idx >> 4;
            int d8  = (idx & 15) << 3;
            v8bf val = *(const v8bf*)(vbase + (size_t)(key0 + key) * kHD + d8);
#pragma unroll
            for (int e = 0; e < 8; ++e)
                Vt[(d8 + e) * VST + key] = val[e];
        }
        __syncthreads();

        // Causal early-out: this wave's rows end at qrow0+15 (wave-uniform
        // branch, EXEC stays all-ones for the WMMA region inside).
        if (key0 <= qrow0 + 15) {
            // ---- scores: S = Q @ K^T  (16 rows x 32 keys) ----
            v8f sc[2];
            sc[0] = v8f{}; sc[1] = v8f{};
#pragma unroll
            for (int nj = 0; nj < 2; ++nj) {
#pragma unroll
                for (int c = 0; c < 4; ++c) {
                    const __bf16* p = &Ks[(nj * 16 + lr) * KST + c * 32 + kg * 8];
                    v8bf lo = *(const v8bf*)p;
                    v8bf hi = *(const v8bf*)(p + 16);
                    v16bf kf = CAT16(lo, hi);
                    sc[nj] = __builtin_amdgcn_wmma_f32_16x16x32_bf16(
                        false, qf[c], false, kf, (short)0, sc[nj], false, false);
                }
            }

            // ---- causal mask + scale (per-lane select, no divergence) ----
            float sv[16];
#pragma unroll
            for (int nj = 0; nj < 2; ++nj)
#pragma unroll
                for (int j = 0; j < 8; ++j) {
                    int row = qrow0 + kg * 8 + j;
                    int key = key0 + nj * 16 + lr;
                    float val = sc[nj][j] * scale;
                    sv[nj * 8 + j] = (key <= row) ? val : -1e30f;
                }

            // ---- online softmax (rows replicated across 16-lane halves) ----
            float pj[16];
#pragma unroll
            for (int j = 0; j < 8; ++j) {
                float rmax = fmaxf(sv[j], sv[8 + j]);
                for (int off = 1; off < 16; off <<= 1)
                    rmax = fmaxf(rmax, __shfl_xor(rmax, off, 16));
                float mn   = fmaxf(m[j], rmax);
                float corr = __expf(m[j] - mn);
                m[j] = mn;
                l[j] *= corr;
#pragma unroll
                for (int t2 = 0; t2 < 8; ++t2) o[t2][j] *= corr;
                float p0 = __expf(sv[j]     - mn);
                float p1 = __expf(sv[8 + j] - mn);
                float ps = p0 + p1;
                for (int off = 1; off < 16; off <<= 1)
                    ps += __shfl_xor(ps, off, 16);
                l[j] += ps;
                pj[j] = p0; pj[8 + j] = p1;
            }

            // ---- relayout P (C-layout -> A-layout) via per-wave LDS tile ----
            __bf16* pp = &Ps[wave][0];
#pragma unroll
            for (int nj = 0; nj < 2; ++nj)
#pragma unroll
                for (int j = 0; j < 8; ++j)
                    pp[(kg * 8 + j) * PST + nj * 16 + lr] = (__bf16)pj[nj * 8 + j];

            v8bf plo = *(const v8bf*)&pp[lr * PST + kg * 8];
            v8bf phi = *(const v8bf*)&pp[lr * PST + 16 + kg * 8];
            v16bf pf = CAT16(plo, phi);

            // ---- out += P @ V  (16 rows x 128 d, K = 32 keys) ----
#pragma unroll
            for (int t2 = 0; t2 < 8; ++t2) {
                const __bf16* vp = &Vt[(t2 * 16 + lr) * VST + kg * 8];
                v8bf vlo = *(const v8bf*)vp;
                v8bf vhi = *(const v8bf*)(vp + 16);
                v16bf vf = CAT16(vlo, vhi);
                o[t2] = __builtin_amdgcn_wmma_f32_16x16x32_bf16(
                    false, pf, false, vf, (short)0, o[t2], false, false);
            }
        }
    }

    // ---- normalize and write bf16 attn output (B, S, HID) ----
#pragma unroll
    for (int t2 = 0; t2 < 8; ++t2) {
        int dcol = t2 * 16 + lr;
#pragma unroll
        for (int j = 0; j < 8; ++j) {
            int row = qrow0 + kg * 8 + j;
            float val = o[t2][j] / l[j];
            O[((size_t)b * kS + row) * kHid + h * kHD + dcol] = (__bf16)val;
        }
    }
}

// ---------------------------------------------------------------------------
// Launcher. Inputs: x, position_ids, mask, w_qkv, w_o.  Output f32 (B,S,HID).
// Workspace layout (72 MB peak, attn aliases the dead qkv region):
//   [0, 48MB)   qkv f32          -> later reused as attn bf16 (16MB)
//   [48, 64MB)  q bf16
//   [64, 68MB)  k bf16
//   [68, 72MB)  v bf16
// ---------------------------------------------------------------------------
extern "C" void kernel_launch(void* const* d_in, const int* in_sizes, int n_in,
                              void* d_out, int out_size, void* d_ws, size_t ws_size,
                              hipStream_t stream)
{
    (void)in_sizes; (void)n_in; (void)out_size; (void)ws_size;

    const float* x     = (const float*)d_in[0];
    const float* w_qkv = (const float*)d_in[3];
    const float* w_o   = (const float*)d_in[4];
    float* out = (float*)d_out;

    char* ws = (char*)d_ws;
    float*  qkv  = (float*)ws;
    __bf16* qb   = (__bf16*)(ws + (size_t)kMRows * kQKVD * sizeof(float));
    __bf16* kb   = qb + (size_t)kB * kNH  * kS * kHD;
    __bf16* vb   = kb + (size_t)kB * kNKV * kS * kHD;
    __bf16* attn = (__bf16*)ws;    // aliases qkv (dead after rope kernel)

    dim3 blk(256);

    // 1) qkv = x @ w_qkv^T      (M=4096, N=3072, K=2048)
    gemm_bf16_wmma<false><<<dim3(kQKVD / 128, kMRows / 128), blk, 0, stream>>>(
        x, w_qkv, qkv, kMRows, kQKVD, kHid);

    // 2) RoPE + split to head-major bf16
    rope_split_kernel<<<dim3(kMRows), blk, 0, stream>>>(qkv, qb, kb, vb);

    // 3) causal flash attention (GQA)
    flash_attn_kernel<<<dim3(kS / 128, kNH, kB), blk, 0, stream>>>(qb, kb, vb, attn);

    // 4) out = attn @ w_o^T     (M=4096, N=2048, K=2048)
    gemm_bf16_wmma<true><<<dim3(kHid / 128, kMRows / 128), blk, 0, stream>>>(
        attn, w_o, out, kMRows, kHid, kHid);
}
